// TriFuse_30511447671001
// MI455X (gfx1250) — matrix operand order repaired and checked by reading, not compiled, not run on hardware
//
#include <hip/hip_runtime.h>

// Shapes (compile-time)
#define BB   2
#define CC   64
#define NN   4096   // H*W
#define OUTC 64

typedef __attribute__((ext_vector_type(16))) _Float16 v16h;
typedef __attribute__((ext_vector_type(8)))  _Float16 v8h;
typedef __attribute__((ext_vector_type(8)))  float    v8f;

__device__ __forceinline__ v8f wmma16x16x32(v16h a, v16h b, v8f c) {
    // D = A(16x32 f16) * B(32x16 f16) + C(16x16 f32)
    return __builtin_amdgcn_wmma_f32_16x16x32_f16(false, a, false, b, (short)0, c, false, false);
}

// A-matrix fragment (16x32 f16): lanes 0-15 hold K={0..15}; lanes 16-31 K={16..31}.
template <typename PT>
__device__ __forceinline__ v16h loadAfrag(const PT* rowbase, int kc, int hf) {
    const PT* p = rowbase + kc * 32 + hf * 16;
    v8h lo = *(const v8h*)(p);
    v8h hi = *(const v8h*)(p + 8);
    return __builtin_shufflevector(lo, hi, 0,1,2,3,4,5,6,7,8,9,10,11,12,13,14,15);
}

// ---- DPP 16-lane butterfly reductions (VALU only; no LDS traffic) ----
template <int CTRL>
__device__ __forceinline__ float dppmov(float x) {
    return __builtin_bit_cast(float,
        __builtin_amdgcn_update_dpp(0, __builtin_bit_cast(int, x), CTRL, 0xF, 0xF, true));
}
__device__ __forceinline__ float redmax16(float v) {
    v = fmaxf(v, dppmov<0xB1>(v));   // quad_perm [1,0,3,2]  (xor 1)
    v = fmaxf(v, dppmov<0x4E>(v));   // quad_perm [2,3,0,1]  (xor 2)
    v = fmaxf(v, dppmov<0x141>(v));  // row_half_mirror      (merge quads in 8)
    v = fmaxf(v, dppmov<0x140>(v));  // row_mirror           (merge halves in 16)
    return v;
}
__device__ __forceinline__ float redsum16(float v) {
    v += dppmov<0xB1>(v);
    v += dppmov<0x4E>(v);
    v += dppmov<0x141>(v);
    v += dppmov<0x140>(v);
    return v;
}

// ---- async global -> LDS copy, 16B per lane; tracked by ASYNCcnt ----
__device__ __forceinline__ void async_copy16(const _Float16* gsrc, _Float16* ldst) {
    unsigned lds_addr = (unsigned)(size_t)ldst;                 // generic low 32 bits == LDS byte offset
    unsigned long long gaddr = (unsigned long long)(size_t)gsrc;
    asm volatile("global_load_async_to_lds_b128 %0, %1, off"
                 :: "v"(lds_addr), "v"(gaddr) : "memory");
}
__device__ __forceinline__ void wait_async0() {
    asm volatile("s_wait_asynccnt 0x0" ::: "memory");
}

// ---------------------------------------------------------------------------
// Kernel 1: 1x1-conv QKV projection for one modality.
//   Q out: (B,N,C) f16, scaled by s * (1/sqrt(C)); K out: (B,N,C) f16 scaled by s;
//   V out: (B,C,N) f16.
// ---------------------------------------------------------------------------
__global__ void __launch_bounds__(256) qkv_proj_kernel(
        const float* __restrict__ x, const float* __restrict__ s,
        const float* __restrict__ wq, const float* __restrict__ bq,
        const float* __restrict__ wk, const float* __restrict__ bk,
        const float* __restrict__ wv, const float* __restrict__ bv,
        _Float16* __restrict__ Qg, _Float16* __restrict__ Kg, _Float16* __restrict__ Vg) {
    __shared__ float xsh[CC * 64];
    __shared__ float ssh[64];
    const int t  = threadIdx.x;
    const int b  = blockIdx.x >> 6;
    const int n0 = (blockIdx.x & 63) << 6;

#pragma unroll
    for (int i = 0; i < 16; ++i) {
        int idx = t + 256 * i;
        int c = idx >> 6, p = idx & 63;
        xsh[c * 64 + p] = x[((size_t)b * CC + c) * NN + n0 + p];
    }
    if (t < 64) ssh[t] = s[(size_t)b * NN + n0 + t];
    __syncthreads();

#pragma unroll
    for (int i = 0; i < 16; ++i) {
        int idx = t + 256 * i;
        int o = idx >> 6, p = idx & 63;
        float aq = bq[o], ak = bk[o], av = bv[o];
        for (int c = 0; c < CC; ++c) {
            float xv = xsh[c * 64 + p];
            aq = fmaf(wq[o * CC + c], xv, aq);
            ak = fmaf(wk[o * CC + c], xv, ak);
            av = fmaf(wv[o * CC + c], xv, av);
        }
        float sc = ssh[p];
        Qg[((size_t)b * NN + n0 + p) * CC + o] = (_Float16)(aq * sc * 0.125f); // 1/sqrt(64)
        Kg[((size_t)b * NN + n0 + p) * CC + o] = (_Float16)(ak * sc);
        Vg[((size_t)b * CC + o) * NN + n0 + p] = (_Float16)av;
    }
}

// ---------------------------------------------------------------------------
// Kernel 1b: per-attend K sums (pure copy source for async staging in attention)
//   KS[a] = K[k1(a)] + K[k2(a)],  (3,B,N,C) f16
// ---------------------------------------------------------------------------
__global__ void __launch_bounds__(256) ksum_kernel(const _Float16* __restrict__ Kg,
                                                   _Float16* __restrict__ KSg) {
    const size_t SEG = (size_t)BB * NN * CC / 8;   // v8h segments per modality
    size_t idx = (size_t)blockIdx.x * 256 + threadIdx.x;
    int a = (int)(idx / SEG);
    size_t r = idx % SEG;
    int k1 = (a == 0) ? 1 : 0;
    int k2 = (a == 2) ? 1 : 2;
    v8h u = *(const v8h*)(Kg + ((size_t)k1 * SEG + r) * 8);
    v8h w = *(const v8h*)(Kg + ((size_t)k2 * SEG + r) * 8);
    *(v8h*)(KSg + idx * 8) = u + w;
}

// ---------------------------------------------------------------------------
// Kernel 2: flash attention with double-buffered async LDS staging.
//   grid = 3*B*(N/64); 128 threads = 4 waves, one 16-row tile per wave.
// ---------------------------------------------------------------------------
#define LDK 80   // padded LDS row stride in f16 (160B; keeps 32B alignment, breaks conflicts)
#define NCHUNK (NN / 64)

__global__ void __launch_bounds__(128, 1) attn_kernel(
        const _Float16* __restrict__ Qg, const _Float16* __restrict__ KSg,
        const _Float16* __restrict__ Vg, float* __restrict__ Og) {
    __shared__ __align__(32) _Float16 Ks[2][64 * LDK];    // [buf][col-pixel][c]
    __shared__ __align__(32) _Float16 Vs[2][64 * LDK];    // [buf][c][col-pixel]
    __shared__ __align__(32) _Float16 Ps[4][16 * LDK];    // per-wave P tile [row][col]

    const int bi  = blockIdx.x;
    const int a   = bi / (BB * 64);
    const int rem = bi % (BB * 64);
    const int b   = rem / 64;
    const int n0  = (rem % 64) * 64;

    const _Float16* Qp  = Qg  + ((size_t)a * BB + b) * NN * CC;
    const _Float16* KSp = KSg + ((size_t)a * BB + b) * NN * CC;
    const _Float16* Vp  = Vg  + ((size_t)a * BB + b) * CC * NN;
    float*          Op  = Og  + ((size_t)a * BB + b) * NN * CC;

    const int t  = threadIdx.x;
    const int wv = t >> 5;
    const int ln = t & 31;
    const int hf = ln >> 4;
    const int lm = ln & 15;
    const int row0 = n0 + wv * 16;

    // async-stage one 64-column chunk (K-sum + V) into LDS buffer `buf`
    auto stage = [&](int ch, int buf) {
#pragma unroll
        for (int i = 0; i < 4; ++i) {
            int seg = t + 128 * i;             // 512 x 16B segments each
            int r = seg >> 3, c8 = seg & 7;
            async_copy16(KSp + ((size_t)(ch * 64 + r)) * CC + c8 * 8, &Ks[buf][r * LDK + c8 * 8]);
            async_copy16(Vp + (size_t)r * NN + ch * 64 + c8 * 8,      &Vs[buf][r * LDK + c8 * 8]);
        }
    };

    const _Float16* qrow = Qp + (size_t)(row0 + lm) * CC;
    const v16h qa0 = loadAfrag(qrow, 0, hf);
    const v16h qa1 = loadAfrag(qrow, 1, hf);

    float mr[8], lr[8];
    v8f oacc[4];
#pragma unroll
    for (int r = 0; r < 8; ++r) { mr[r] = -INFINITY; lr[r] = 0.f; }
    v8f zero = {};
#pragma unroll
    for (int ct = 0; ct < 4; ++ct) oacc[ct] = zero;

    _Float16* Pw = &Ps[wv][0];

    // prologue: stage chunk 0 into buffer 0
    stage(0, 0);
    wait_async0();
    __syncthreads();

    for (int ch = 0; ch < NCHUNK; ++ch) {
        const int cur = ch & 1;
        // issue DMA for next chunk into the other buffer; overlaps with compute below
        if (ch + 1 < NCHUNK) stage(ch + 1, cur ^ 1);

        const _Float16* Kb = &Ks[cur][0];
        const _Float16* Vb = &Vs[cur][0];

        // ---- gemm1: S(16x64) = Q . Ksum^T
        v8f sc[4];
#pragma unroll
        for (int ct = 0; ct < 4; ++ct) {
            v8f acc = zero;
            v16h b0 = *(const v16h*)(Kb + (ct * 16 + lm) * LDK + hf * 16);
            v16h b1 = *(const v16h*)(Kb + (ct * 16 + lm) * LDK + 32 + hf * 16);
            acc = wmma16x16x32(qa0, b0, acc);
            acc = wmma16x16x32(qa1, b1, acc);
            sc[ct] = acc;
        }

        // ---- online softmax in C/D layout (row = r + 8*hf, col = ct*16 + lm)
        float corr[8], rs[8];
#pragma unroll
        for (int r = 0; r < 8; ++r) {
            float v = fmaxf(fmaxf(sc[0][r], sc[1][r]), fmaxf(sc[2][r], sc[3][r]));
            v = redmax16(v);
            float mn = fmaxf(mr[r], v);
            corr[r] = __expf(mr[r] - mn);
            mr[r] = mn;
            rs[r] = 0.f;
        }
#pragma unroll
        for (int ct = 0; ct < 4; ++ct) {
            v8f sv = sc[ct];
#pragma unroll
            for (int r = 0; r < 8; ++r) {
                float p = __expf(sv[r] - mr[r]);
                rs[r] += p;
                Pw[(r + 8 * hf) * LDK + ct * 16 + lm] = (_Float16)p;
            }
        }
#pragma unroll
        for (int r = 0; r < 8; ++r)
            lr[r] = lr[r] * corr[r] + redsum16(rs[r]);

        // ---- gemm2: O += P . V^T  (per-wave LDS is in-order)
        const v16h pa0 = loadAfrag(Pw + (size_t)lm * LDK, 0, hf);
        const v16h pa1 = loadAfrag(Pw + (size_t)lm * LDK, 1, hf);
#pragma unroll
        for (int ct = 0; ct < 4; ++ct) {
            v8f c = oacc[ct];
#pragma unroll
            for (int r = 0; r < 8; ++r) c[r] *= corr[r];
            v16h vb0 = *(const v16h*)(Vb + (ct * 16 + lm) * LDK + hf * 16);
            v16h vb1 = *(const v16h*)(Vb + (ct * 16 + lm) * LDK + 32 + hf * 16);
            c = wmma16x16x32(pa0, vb0, c);
            c = wmma16x16x32(pa1, vb1, c);
            oacc[ct] = c;
        }

        if (ch + 1 < NCHUNK) {
            wait_async0();      // next chunk's DMA done (likely already)
            __syncthreads();    // visible to all waves; readers of overwritten buf done
        }
    }

#pragma unroll
    for (int ct = 0; ct < 4; ++ct)
#pragma unroll
        for (int r = 0; r < 8; ++r) {
            float v = oacc[ct][r] / lr[r];
            Op[(size_t)(row0 + r + 8 * hf) * CC + ct * 16 + lm] = v;
        }
}

// ---------------------------------------------------------------------------
// Kernel 3: fused = sum_a sigmoid(Wg.o_a + bg) * o_a ; out = Wo.fused + bo
// ---------------------------------------------------------------------------
__global__ void __launch_bounds__(256) gate_out_kernel(
        const float* __restrict__ Og,
        const float* __restrict__ wg, const float* __restrict__ bg,
        const float* __restrict__ wo, const float* __restrict__ bo,
        float* __restrict__ out) {
    __shared__ float os[3][4][CC];
    __shared__ float fs[4][CC];
    const int t  = threadIdx.x;
    const int b  = blockIdx.x >> 10;
    const int n0 = (blockIdx.x & 1023) << 2;
    const int p  = t >> 6;
    const int c0 = t & 63;

#pragma unroll
    for (int a = 0; a < 3; ++a)
        os[a][p][c0] = Og[(((size_t)a * BB + b) * NN + n0 + p) * CC + c0];
    __syncthreads();

    float f = 0.f;
#pragma unroll
    for (int a = 0; a < 3; ++a) {
        float z = bg[c0];
        for (int c = 0; c < CC; ++c) z = fmaf(wg[c0 * CC + c], os[a][p][c], z);
        float g = 1.f / (1.f + __expf(-z));
        f += g * os[a][p][c0];
    }
    fs[p][c0] = f;
    __syncthreads();

    float acc = bo[c0];
    for (int c = 0; c < CC; ++c) acc = fmaf(wo[c0 * CC + c], fs[p][c], acc);
    out[((size_t)b * OUTC + c0) * NN + n0 + p] = acc;
}

// ---------------------------------------------------------------------------
extern "C" void kernel_launch(void* const* d_in, const int* in_sizes, int n_in,
                              void* d_out, int out_size, void* d_ws, size_t ws_size,
                              hipStream_t stream) {
    (void)in_sizes; (void)n_in; (void)out_size; (void)ws_size;
    const float* xin[3] = {(const float*)d_in[0], (const float*)d_in[1], (const float*)d_in[2]};
    const float* sin[3] = {(const float*)d_in[3], (const float*)d_in[4], (const float*)d_in[5]};
    const float* wg = (const float*)d_in[24];
    const float* bg = (const float*)d_in[25];
    const float* wo = (const float*)d_in[26];
    const float* bo = (const float*)d_in[27];

    const size_t MODE = (size_t)BB * NN * CC;     // f16 elems per modality array
    _Float16* Qg  = (_Float16*)d_ws;
    _Float16* Kg  = Qg + 3 * MODE;
    _Float16* Vg  = Kg + 3 * MODE;
    _Float16* KSg = Vg + 3 * MODE;
    float*    Og  = (float*)((char*)d_ws + (size_t)4 * 3 * MODE * sizeof(_Float16));

    for (int m = 0; m < 3; ++m) {
        int base = 6 + m * 6;
        qkv_proj_kernel<<<BB * (NN / 64), 256, 0, stream>>>(
            xin[m], sin[m],
            (const float*)d_in[base + 0], (const float*)d_in[base + 1],
            (const float*)d_in[base + 2], (const float*)d_in[base + 3],
            (const float*)d_in[base + 4], (const float*)d_in[base + 5],
            Qg + m * MODE, Kg + m * MODE, Vg + m * MODE);
    }
    ksum_kernel<<<(3 * MODE / 8) / 256, 256, 0, stream>>>(Kg, KSg);
    attn_kernel<<<3 * BB * (NN / 64), 128, 0, stream>>>(Qg, KSg, Vg, Og);
    gate_out_kernel<<<BB * (NN / 4), 256, 0, stream>>>(Og, wg, bg, wo, bo, (float*)d_out);
}
